// GraphSage_31825707663803
// MI455X (gfx1250) — matrix operand order repaired and checked by reading, not compiled, not run on hardware
//
#include <hip/hip_runtime.h>
#include <hip/hip_bf16.h>

// Problem constants (from reference): N=100000, D=256, OUT=256, K=10, B=8192
#define NODES 100000
#define DFEAT 256
#define DOUT  256
#define KNB   10
#define BATCH 8192
#define TWOD  512   // 2*D == 2*OUT

// LDS plan for the GEMM: full W tile + double-buffered 32-row A tile.
#define WLDS_BYTES (DOUT * TWOD * 2)              // 262144 (256 KB)
#define ALDS_BYTES (32 * TWOD * 2)                // 32768  (32 KB)
#define SMEM_BYTES (WLDS_BYTES + 2 * ALDS_BYTES)  // 327680 = 320 KB exactly

typedef __attribute__((ext_vector_type(16))) __bf16 bf16x16;
typedef __attribute__((ext_vector_type(8)))  float  f32x8;

// Scheduler pinning: [DS-read x10 (next iter's preloads)] [WMMA x4 (this iter)]
#if __has_builtin(__builtin_amdgcn_sched_group_barrier)
#define SCHED_GROUP(mask, n, id) __builtin_amdgcn_sched_group_barrier(mask, n, id)
#else
#define SCHED_GROUP(mask, n, id)
#endif
#define SGB_DSREAD 0x0100
#define SGB_MFMA   0x0008

// ---------------------------------------------------------------------------
// Fragment-native ("swizzled") storage order.
// A 1 KB fragment block = 32 lanes x 32 contiguous bytes (16 bf16 per lane).
// GEMM-side reads are then lane*32 + const -> conflict-free LDS access, and
// async LDS fills are pure linear copies.
//
// A-fragment (16x32, ISA 7.12.2): lane (m=l&15, half=l>>4) holds
//   A[m][ks*32 + half*8 .. +7]  ++  A[m][ks*32 + 16 + half*8 .. +7]
// B-fragment (32x16): lane (n=l&15, half=l>>4) holds
//   W[n][ks*32 + half*8 .. +7]  ++  W[n][ks*32 + 16 + half*8 .. +7]
// ---------------------------------------------------------------------------
__device__ __forceinline__ int a_swz_pos(int rw, int kcol) {
  const int rt   = rw >> 4;              // row tile 0..1
  const int l16r = rw & 15;
  const int ks   = kcol >> 5;            // k-step 0..15
  const int r    = kcol & 31;
  const int half = (r >> 3) & 1;         // 8-wide interleave
  const int e    = (r & 7) + ((r >> 4) << 3);
  const int lane = half * 16 + l16r;
  return (((rt * 16 + ks) * 32) + lane) * 16 + e;
}
__device__ __forceinline__ int w_swz_pos(int o, int k) {
  const int jt   = o >> 4;               // col tile 0..15
  const int l16r = o & 15;
  const int ks   = k >> 5;
  const int r    = k & 31;
  const int half = (r >> 3) & 1;         // 8-wide interleave
  const int e    = (r & 7) + ((r >> 4) << 3);
  const int lane = half * 16 + l16r;
  return (((jt * 16 + ks) * 32) + lane) * 16 + e;
}

// ---------------------------------------------------------------------------
// CDNA5 async global->LDS copy (ASYNCcnt-tracked, ISA "Async LDS <-> Memory")
// ---------------------------------------------------------------------------
__device__ __forceinline__ void async_copy_b128(uint32_t lds_off, uint64_t gaddr) {
  asm volatile("global_load_async_to_lds_b128 %0, %1, off"
               :: "v"(lds_off), "v"(gaddr) : "memory");
}
__device__ __forceinline__ void wait_async0() {
  asm volatile("s_wait_asynccnt 0x0" ::: "memory");
}

// ---------------------------------------------------------------------------
// W (fp32 [OUT, 2D]) -> bf16, fragment-swizzled
// ---------------------------------------------------------------------------
__global__ __launch_bounds__(256) void cvt_w_kernel(const float* __restrict__ W,
                                                    __bf16* __restrict__ WbSwz, int n) {
  int i = blockIdx.x * 256 + threadIdx.x;
  if (i < n) {
    const int o = i >> 9;          // /512
    const int k = i & 511;
    WbSwz[w_swz_pos(o, k)] = (__bf16)W[i];
  }
}

// ---------------------------------------------------------------------------
// Layer-1 input build: fuses L1 normalization, K=10 gather, weighted mean,
// concat, cvt->bf16, and the fragment swizzle. One block per node.
// ---------------------------------------------------------------------------
__global__ __launch_bounds__(256) void build1_kernel(const float* __restrict__ feat,
                                                     const int*   __restrict__ nidx,
                                                     const float* __restrict__ ec,
                                                     __bf16*      __restrict__ ASwz) {
  __shared__ float w[KNB];
  __shared__ int   ng[KNB];
  const int n = blockIdx.x;
  const int d = threadIdx.x;
  if (d < KNB) {
    w[d]  = ec[(size_t)n * KNB + d];
    ng[d] = nidx[(size_t)n * KNB + d];
  }
  __syncthreads();
  float s = 0.f;
#pragma unroll
  for (int k = 0; k < KNB; ++k) s += w[k];
  const float inv = 1.f / fmaxf(s, 1e-12f);

  const float self = feat[(size_t)n * DFEAT + d];
  float agg = 0.f;
#pragma unroll
  for (int k = 0; k < KNB; ++k) agg += w[k] * feat[(size_t)ng[k] * DFEAT + d];

  __bf16* blkp = ASwz + (size_t)(n >> 5) * (32 * TWOD);
  const int rw = n & 31;
  blkp[a_swz_pos(rw, d)]         = (__bf16)self;
  blkp[a_swz_pos(rw, DFEAT + d)] = (__bf16)(agg * inv);
}

// ---------------------------------------------------------------------------
// Layer-2 input build: only the B=8192 batch nodes; gathers from bf16 h1.
// ---------------------------------------------------------------------------
__global__ __launch_bounds__(256) void build2_kernel(const __bf16* __restrict__ H1,
                                                     const int*    __restrict__ nidx,
                                                     const float*  __restrict__ ec,
                                                     const int*    __restrict__ batch,
                                                     __bf16*       __restrict__ ASwz) {
  __shared__ float w[KNB];
  __shared__ int   ng[KNB];
  __shared__ int   node_s;
  const int b = blockIdx.x;
  const int d = threadIdx.x;
  if (d == 0) node_s = batch[b];
  __syncthreads();
  const int node = node_s;
  if (d < KNB) {
    w[d]  = ec[(size_t)node * KNB + d];
    ng[d] = nidx[(size_t)node * KNB + d];
  }
  __syncthreads();
  float s = 0.f;
#pragma unroll
  for (int k = 0; k < KNB; ++k) s += w[k];
  const float inv = 1.f / fmaxf(s, 1e-12f);

  const float self = (float)H1[(size_t)node * DOUT + d];
  float agg = 0.f;
#pragma unroll
  for (int k = 0; k < KNB; ++k) agg += w[k] * (float)H1[(size_t)ng[k] * DOUT + d];

  __bf16* blkp = ASwz + (size_t)(b >> 5) * (32 * TWOD);
  const int rw = b & 31;
  blkp[a_swz_pos(rw, d)]         = (__bf16)self;
  blkp[a_swz_pos(rw, DFEAT + d)] = (__bf16)(agg * inv);
}

// ---------------------------------------------------------------------------
// Persistent WMMA GEMM + ReLU:  C[M,256] = relu(A[M,512] @ W[256,512]^T)
// A and W arrive fragment-swizzled. 8 waves = 2 (row) x 4 (col); each wave
// owns a 16x64 strip -> 4 accumulators. Per k-step: 1 A frag + 4 B frags,
// each a single contiguous 32 B LDS read at lane*32 (bank-conflict-free).
// sched_group_barrier pins each unrolled step to [10 DS-read][4 WMMA] so the
// loads belong to iteration k+1 while the WMMAs consume iteration k ->
// partial dscnt waits instead of full drains. Async fills (ASYNCcnt)
// prefetch the next 32-row A block during the current block's WMMA loop.
// ---------------------------------------------------------------------------
template <typename OutT>
__global__ __launch_bounds__(256) void gemm_relu_kernel(const __bf16* __restrict__ ASwz,
                                                        const __bf16* __restrict__ WbSwz,
                                                        OutT* __restrict__ C,
                                                        int nblk) {
  extern __shared__ char smem[];
  const uint32_t lds0 = (uint32_t)(uintptr_t)smem;   // LDS offset of smem[0]

  const int tid  = threadIdx.x;
  const int wid  = tid >> 5;
  const int lane = tid & 31;
  const int l16  = lane & 15;
  const int half = lane >> 4;
  const int wrow = wid >> 2;   // 0..1
  const int wcol = wid & 3;    // 0..3
  const int cb   = wcol * 64;  // output col base
  const int rb0  = blockIdx.x * nblk;

  // ---- async fills are pure linear copies (data pre-swizzled in global) ----
  for (int i = tid; i < WLDS_BYTES / 16; i += 256)
    async_copy_b128(lds0 + (uint32_t)i * 16,
                    (uint64_t)(uintptr_t)WbSwz + (uint64_t)i * 16);

  auto fillA = [&](int rb, int buf) {
    const uint64_t g = (uint64_t)(uintptr_t)ASwz + (uint64_t)rb * ALDS_BYTES;
    const uint32_t l = lds0 + WLDS_BYTES + (uint32_t)buf * ALDS_BYTES;
#pragma unroll
    for (int i = tid; i < ALDS_BYTES / 16; i += 256)   // 8 chunks / thread
      async_copy_b128(l + (uint32_t)i * 16, g + (uint64_t)i * 16);
  };
  fillA(rb0, 0);
  wait_async0();
  __syncthreads();

  // per-wave fragment base pointers (lane*32 within each 1 KB fragment block)
  const char* Bb = smem + (size_t)wcol * (4 * 16 * 1024) + lane * 32;

  for (int b = 0; b < nblk; ++b) {
    const int rb = rb0 + b;
    if (b + 1 < nblk) fillA(rb0 + b + 1, (b + 1) & 1);   // overlaps WMMA loop

    const char* Ab = smem + WLDS_BYTES + (b & 1) * ALDS_BYTES
                   + (size_t)wrow * (16 * 1024) + lane * 32;

    f32x8 acc0 = {}, acc1 = {}, acc2 = {}, acc3 = {};

    // ---- software-pipelined K loop: 16 steps, frag offsets = ks*1024 ----
    bf16x16 aC  = *(const bf16x16*)(Ab);
    bf16x16 b0C = *(const bf16x16*)(Bb);
    bf16x16 b1C = *(const bf16x16*)(Bb + 16384);
    bf16x16 b2C = *(const bf16x16*)(Bb + 32768);
    bf16x16 b3C = *(const bf16x16*)(Bb + 49152);
#pragma unroll
    for (int ks = 0; ks < 16; ++ks) {
      const int kn = (ks + 1) & 15;   // wrap: last-iter preloads are dead
      bf16x16 aN  = *(const bf16x16*)(Ab + kn * 1024);
      bf16x16 b0N = *(const bf16x16*)(Bb + kn * 1024);
      bf16x16 b1N = *(const bf16x16*)(Bb + 16384 + kn * 1024);
      bf16x16 b2N = *(const bf16x16*)(Bb + 32768 + kn * 1024);
      bf16x16 b3N = *(const bf16x16*)(Bb + 49152 + kn * 1024);

      acc0 = __builtin_amdgcn_wmma_f32_16x16x32_bf16(false, aC, false, b0C, (short)0, acc0, false, false);
      acc1 = __builtin_amdgcn_wmma_f32_16x16x32_bf16(false, aC, false, b1C, (short)0, acc1, false, false);
      acc2 = __builtin_amdgcn_wmma_f32_16x16x32_bf16(false, aC, false, b2C, (short)0, acc2, false, false);
      acc3 = __builtin_amdgcn_wmma_f32_16x16x32_bf16(false, aC, false, b3C, (short)0, acc3, false, false);

      // Pin the schedule: this iteration's 10 DS reads (k+1 preloads) are
      // emitted BEFORE the 4 WMMAs that consume iteration k's fragments.
      SCHED_GROUP(SGB_DSREAD, 10, 0);
      SCHED_GROUP(SGB_MFMA,    4, 0);

      aC = aN; b0C = b0N; b1C = b1N; b2C = b2N; b3C = b3N;
    }

    // ---- fused ReLU + store (C layout: VGPR r -> row half*8+r, col l16) ----
    const int tr = rb * 32 + wrow * 16;
#pragma unroll
    for (int r = 0; r < 8; ++r) {
      const size_t crow = (size_t)(tr + half * 8 + r) * DOUT;
      C[crow + cb +  0 + l16] = (OutT)fmaxf(acc0[r], 0.f);
      C[crow + cb + 16 + l16] = (OutT)fmaxf(acc1[r], 0.f);
      C[crow + cb + 32 + l16] = (OutT)fmaxf(acc2[r], 0.f);
      C[crow + cb + 48 + l16] = (OutT)fmaxf(acc3[r], 0.f);
    }

    // ensure next A buffer is filled everywhere, and that no wave still reads
    // buf[b&1] before it is overwritten in iteration b+1.
    wait_async0();
    __syncthreads();
  }
}

// ---------------------------------------------------------------------------
// kernel_launch: inputs in setup_inputs() order:
//   0 features[N,D] f32, 1 neigh_idx[N,K] i32, 2 edge_counts[N,K] f32,
//   3 W1[OUT,2D] f32, 4 W2[OUT,2OUT] f32, 5 nodes_batch[B] i32
// Output: h2[nodes_batch] -> [B, OUT] f32
// ---------------------------------------------------------------------------
extern "C" void kernel_launch(void* const* d_in, const int* in_sizes, int n_in,
                              void* d_out, int out_size, void* d_ws, size_t ws_size,
                              hipStream_t stream) {
  const float* feat = (const float*)d_in[0];
  const int*   nidx = (const int*)d_in[1];
  const float* ec   = (const float*)d_in[2];
  const float* W1   = (const float*)d_in[3];
  const float* W2   = (const float*)d_in[4];
  const int*   nb   = (const int*)d_in[5];
  float* out = (float*)d_out;

  // Workspace carve-up (~163 MB), 256B-aligned. Every buffer is fully written
  // each call before being read -> deterministic, poison-safe.
  char*  ws  = (char*)d_ws;
  size_t off = 0;
  auto carve = [&](size_t bytes) -> char* {
    char* p = ws + off;
    off += (bytes + 255) & ~(size_t)255;
    return p;
  };
  __bf16* Wb1  = (__bf16*)carve((size_t)DOUT * TWOD * sizeof(__bf16));   // 256 KB (swizzled)
  __bf16* Wb2  = (__bf16*)carve((size_t)DOUT * TWOD * sizeof(__bf16));   // 256 KB (swizzled)
  __bf16* Acat = (__bf16*)carve((size_t)NODES * TWOD * sizeof(__bf16));  // 102.4 MB (swizzled)
  __bf16* H1   = (__bf16*)carve((size_t)NODES * DOUT * sizeof(__bf16));  // 51.2 MB (row-major)
  __bf16* A2   = (__bf16*)carve((size_t)BATCH * TWOD * sizeof(__bf16));  // 8.4 MB (swizzled)
  (void)ws_size; (void)in_sizes; (void)n_in; (void)out_size;

  // Allow 320 KB dynamic LDS for the GEMM kernels (host-side setup, not a
  // stream op -> graph-capture safe; idempotent & deterministic).
  (void)hipFuncSetAttribute(reinterpret_cast<const void*>(gemm_relu_kernel<__bf16>),
                            hipFuncAttributeMaxDynamicSharedMemorySize, SMEM_BYTES);
  (void)hipFuncSetAttribute(reinterpret_cast<const void*>(gemm_relu_kernel<float>),
                            hipFuncAttributeMaxDynamicSharedMemorySize, SMEM_BYTES);

  const int wElems = DOUT * TWOD;  // 131072
  cvt_w_kernel<<<(wElems + 255) / 256, 256, 0, stream>>>(W1, Wb1, wElems);
  cvt_w_kernel<<<(wElems + 255) / 256, 256, 0, stream>>>(W2, Wb2, wElems);

  // Layer 1 (dense over all N nodes): 3125 row-blocks = 625 WGs x 5 blocks
  build1_kernel<<<NODES, 256, 0, stream>>>(feat, nidx, ec, Acat);
  gemm_relu_kernel<__bf16><<<625, 256, SMEM_BYTES, stream>>>(Acat, Wb1, H1, 5);

  // Layer 2 (batch rows only): 256 row-blocks = 64 WGs x 4 blocks
  build2_kernel<<<BATCH, 256, 0, stream>>>(H1, nidx, ec, nb, A2);
  gemm_relu_kernel<float><<<64, 256, SMEM_BYTES, stream>>>(A2, Wb2, out, 4);
}
